// _SelfAttention_1760936591594
// MI455X (gfx1250) — matrix operand order repaired and checked by reading, not compiled, not run on hardware
//
#include <hip/hip_runtime.h>
#include <hip/hip_bf16.h>
#include <cstdint>

// ---------------------------------------------------------------------------
// Self-attention (B=4, C=64, N=4096, dqk=8) — flash-attention style, CDNA5.
//   Kernel 1: QKV 1x1-conv projections (weights in LDS).
//             Q,K stored [b][n][8] f32 ; V stored [b][c][n] f16.
//   Kernel 2: per-wave 16-row query tile; 4 waves/block share j-chunks.
//             V (4KB) + K (1KB) chunks double-buffered in LDS via
//             GLOBAL_LOAD_ASYNC_TO_LDS_B128 (ASYNCcnt), overlapping compute.
//             S^T via V_WMMA_F32_16X16X4_F32 (exact fp32, K=8 split 4+4)
//             online softmax per-lane (+ ds_swizzle SWAPX16 half-combine)
//             O^T += V x P^T via V_WMMA_F32_16X16X32_F16 (f32 accum)
// ---------------------------------------------------------------------------

typedef float   v2f  __attribute__((ext_vector_type(2)));
typedef float   v8f  __attribute__((ext_vector_type(8)));
typedef _Float16 v16h __attribute__((ext_vector_type(16)));

#define BATCH 4
#define CCH   64
#define NTOK  4096
#define DQK   8

// SWAPX16: group-of-32 swizzle, xor_mask=0x10, and_mask=0x1f -> 0x401F
__device__ __forceinline__ float swap16f(float x) {
  return __int_as_float(__builtin_amdgcn_ds_swizzle(__float_as_int(x), 0x401F));
}
__device__ __forceinline__ unsigned swap16u(unsigned x) {
  return (unsigned)__builtin_amdgcn_ds_swizzle((int)x, 0x401F);
}
__device__ __forceinline__ unsigned pack2h(float a, float b) {
  union { _Float16 h[2]; unsigned u; } t;
  t.h[0] = (_Float16)a; t.h[1] = (_Float16)b;   // even K in [15:0]
  return t.u;
}

// ---------------------------------------------------------------------------
// Kernel 1: projections. grid = 64 blocks x 256 threads, one thread per (b,n)
// ---------------------------------------------------------------------------
__global__ __launch_bounds__(256) void qkv_proj_kernel(
    const float* __restrict__ x,
    const float* __restrict__ wq, const float* __restrict__ bq,
    const float* __restrict__ wk, const float* __restrict__ bk,
    const float* __restrict__ wv, const float* __restrict__ bv,
    float* __restrict__ Qf, float* __restrict__ Kf, _Float16* __restrict__ Vh)
{
  __shared__ float sWq[DQK * CCH], sWk[DQK * CCH], sWv[CCH * CCH];
  __shared__ float sBq[DQK], sBk[DQK], sBv[CCH];
  const int t = threadIdx.x;
  for (int i = t; i < DQK * CCH; i += 256) { sWq[i] = wq[i]; sWk[i] = wk[i]; }
  for (int i = t; i < CCH * CCH; i += 256) { sWv[i] = wv[i]; }
  if (t < DQK) { sBq[t] = bq[t]; sBk[t] = bk[t]; }
  if (t < CCH) { sBv[t] = bv[t]; }
  __syncthreads();

  const int idx = blockIdx.x * 256 + t;      // 0 .. B*N-1
  const int b = idx >> 12;
  const int n = idx & (NTOK - 1);

  float xr[CCH];
#pragma unroll
  for (int c = 0; c < CCH; ++c)
    xr[c] = x[((size_t)((b << 6) + c) << 12) + n];

  const size_t qkrow = ((size_t)((b << 12) + n)) << 3;
#pragma unroll
  for (int d = 0; d < DQK; ++d) {
    float aq = sBq[d], ak = sBk[d];
#pragma unroll
    for (int c = 0; c < CCH; ++c) {
      aq = __builtin_fmaf(sWq[d * CCH + c], xr[c], aq);
      ak = __builtin_fmaf(sWk[d * CCH + c], xr[c], ak);
    }
    Qf[qkrow + d] = aq;
    Kf[qkrow + d] = ak;
  }

  for (int o = 0; o < CCH; ++o) {
    float av = sBv[o];
#pragma unroll
    for (int c = 0; c < CCH; ++c)
      av = __builtin_fmaf(sWv[(o << 6) + c], xr[c], av);
    Vh[((size_t)((b << 6) + o) << 12) + n] = (_Float16)av;
  }
}

// ---------------------------------------------------------------------------
// Kernel 2: flash attention. grid = B*(N/64) = 256 blocks x 128 threads.
// Wave w handles query rows i0 = blockTile*64 + w*16 .. +15; all 4 waves
// share double-buffered LDS stages of the (K,V) j-chunk.
// ---------------------------------------------------------------------------
__global__ __launch_bounds__(128) void attn_kernel(
    const float* __restrict__ Qf, const float* __restrict__ Kf,
    const _Float16* __restrict__ Vh,
    const float* __restrict__ x, const float* __restrict__ gamma_p,
    float* __restrict__ out)
{
  __shared__ __align__(16) _Float16 sV[2][CCH][32];  // 2 x 4KB
  __shared__ __align__(16) float    sK[2][32][DQK];  // 2 x 1KB

  const int lane = threadIdx.x & 31;
  const int wave = threadIdx.x >> 5;
  const int b     = blockIdx.x >> 6;
  const int itile = blockIdx.x & 63;
  const int i0    = (itile << 6) + (wave << 4);
  const int m15   = lane & 15;
  const bool hi   = lane >= 16;
  const int d0    = hi ? 2 : 0;
  const int irow  = i0 + m15;

  const float*    kb_ = Kf + (((size_t)(b << 12)) << 3);
  const _Float16* vb  = Vh + ((size_t)(b << 6) << 12);

  // ---- async stage of one 32-j chunk into LDS buffer nb -------------------
  auto stage = [&](int jcn, int nb) {
    // V: this wave's 16 channels, 2 x 16B per lane
    const unsigned ldsV = (unsigned)(uintptr_t)(&sV[nb][0][0]);
#pragma unroll
    for (int k2 = 0; k2 < 2; ++k2) {
      const int t2 = lane + (k2 << 5);          // 0..63
      const int c  = (wave << 4) + (t2 >> 2);
      const int part = t2 & 3;                  // 8 halfs each
      const _Float16* g = vb + (((size_t)c) << 12) + jcn + (part << 3);
      const unsigned laddr = ldsV + (unsigned)((c << 6) + (part << 4));
      asm volatile("global_load_async_to_lds_b128 %0, %1, off"
                   :: "v"(laddr), "v"(g) : "memory");
    }
    // K: waves 0,1 stage 32 x 8 f32, 1 x 16B per lane
    if (wave < 2) {
      const unsigned ldsK = (unsigned)(uintptr_t)(&sK[nb][0][0]);
      const int t2 = (wave << 5) + lane;        // 0..63
      const int j  = t2 >> 1;
      const int part = t2 & 1;                  // 4 floats each
      const float* g = kb_ + (((size_t)(jcn + j)) << 3) + (part << 2);
      const unsigned laddr = ldsK + (unsigned)((j << 5) + (part << 4));
      asm volatile("global_load_async_to_lds_b128 %0, %1, off"
                   :: "v"(laddr), "v"(g) : "memory");
    }
  };

  // ---- Q B-fragments (4x16 f32, K=d): VGPR0: d = hi?2:0 ; VGPR1: +1 -------
  const float* qbase = Qf + (((size_t)(b << 12) + irow) << 3) + d0;
  const v2f qf0 = *(const v2f*)(qbase);       // d-chunk 0..3
  const v2f qf1 = *(const v2f*)(qbase + 4);   // d-chunk 4..7

  v8f Oacc[4];
#pragma unroll
  for (int t2 = 0; t2 < 4; ++t2) Oacc[t2] = (v8f){};
  float mrow = -__builtin_inff();
  float lrow = 0.f;

  stage(0, 0);   // prologue: chunk 0 -> buffer 0

  for (int jc = 0; jc < NTOK; jc += 32) {
    const int buf = (jc >> 5) & 1;
    asm volatile("s_wait_asynccnt 0x0" ::: "memory");
    __syncthreads();
    if (jc + 32 < NTOK) stage(jc + 32, buf ^ 1);   // overlap next copy

    // ---- K A-fragments from LDS (16x4 f32, M=j) ---------------------------
    const v2f kA0 = { sK[buf][m15][d0],          sK[buf][m15][d0 + 1] };
    const v2f kA1 = { sK[buf][m15][d0 + 4],      sK[buf][m15][d0 + 5] };
    const v2f kB0 = { sK[buf][16 + m15][d0],     sK[buf][16 + m15][d0 + 1] };
    const v2f kB1 = { sK[buf][16 + m15][d0 + 4], sK[buf][16 + m15][d0 + 5] };

    // ---- S^T tiles: D[j][i] = sum_d K[j,d] * Q[i,d], exact fp32 -----------
    const v8f z = (v8f){};
    v8f sA = __builtin_amdgcn_wmma_f32_16x16x4_f32(false, kA0, false, qf0, (short)0, z,  false, false);
    sA     = __builtin_amdgcn_wmma_f32_16x16x4_f32(false, kA1, false, qf1, (short)0, sA, false, false);
    v8f sB = __builtin_amdgcn_wmma_f32_16x16x4_f32(false, kB0, false, qf0, (short)0, z,  false, false);
    sB     = __builtin_amdgcn_wmma_f32_16x16x4_f32(false, kB1, false, qf1, (short)0, sB, false, false);

    // ---- online softmax: lane owns query column i = irow ------------------
    float cm = sA[0];
#pragma unroll
    for (int r = 1; r < 8; ++r) cm = fmaxf(cm, sA[r]);
#pragma unroll
    for (int r = 0; r < 8; ++r) cm = fmaxf(cm, sB[r]);
    cm = fmaxf(cm, swap16f(cm));                 // combine half-wave j-groups
    const float mnew  = fmaxf(mrow, cm);
    const float scale = __expf(mrow - mnew);     // first iter: exp(-inf) = 0

    float pA[8], pB[8], ps = 0.f;
#pragma unroll
    for (int r = 0; r < 8; ++r) {
      pA[r] = __expf(sA[r] - mnew);
      pB[r] = __expf(sB[r] - mnew);
      ps += pA[r] + pB[r];
    }
    ps += swap16f(ps);
    lrow = lrow * scale + ps;
    mrow = mnew;

#pragma unroll
    for (int t2 = 0; t2 < 4; ++t2)
#pragma unroll
      for (int r = 0; r < 8; ++r) Oacc[t2][r] *= scale;

    // ---- build P^T B-fragment (32x16 f16): VGPRr low: K=2r,2r+1; hi: +16 --
    union { v16h v; unsigned u[8]; } bfr;
#pragma unroll
    for (int r = 0; r < 4; ++r) {
      const unsigned uA = pack2h(pA[2 * r], pA[2 * r + 1]);
      const unsigned uB = pack2h(pB[2 * r], pB[2 * r + 1]);
      const unsigned xA = swap16u(uA);
      const unsigned xB = swap16u(uB);
      bfr.u[r]     = hi ? xB : uA;
      bfr.u[r + 4] = hi ? uB : xA;
    }

    // ---- O^T[c][i] += V[c][j] * P^T[j][i], 4 channel tiles from LDS -------
#pragma unroll
    for (int ct = 0; ct < 4; ++ct) {
      const _Float16* vp = &sV[buf][(ct << 4) + m15][hi ? 16 : 0];
      union { v16h v; uint4 q[2]; } af;   // A 16x32 f16: VGPR0-3 / VGPR4-7
      af.q[0] = *(const uint4*)(vp);
      af.q[1] = *(const uint4*)(vp + 8);
      Oacc[ct] = __builtin_amdgcn_wmma_f32_16x16x32_f16(
          false, af.v, false, bfr.v, (short)0, Oacc[ct], false, false);
    }
  }

  // ---- epilogue: out = gamma * O/l + x -------------------------------------
  const float g   = gamma_p[0];
  const float inv = 1.0f / lrow;
#pragma unroll
  for (int ct = 0; ct < 4; ++ct) {
#pragma unroll
    for (int r = 0; r < 8; ++r) {
      const int c = (ct << 4) + r + (hi ? 8 : 0);
      const size_t oidx = (((size_t)((b << 6) + c)) << 12) + irow;
      out[oidx] = g * (Oacc[ct][r] * inv) + x[oidx];
    }
  }
}

// ---------------------------------------------------------------------------
extern "C" void kernel_launch(void* const* d_in, const int* in_sizes, int n_in,
                              void* d_out, int out_size, void* d_ws, size_t ws_size,
                              hipStream_t stream) {
  const float* x     = (const float*)d_in[0];
  const float* wq    = (const float*)d_in[1];
  const float* bq    = (const float*)d_in[2];
  const float* wk    = (const float*)d_in[3];
  const float* bk    = (const float*)d_in[4];
  const float* wv    = (const float*)d_in[5];
  const float* bv    = (const float*)d_in[6];
  const float* gamma = (const float*)d_in[7];
  float* out = (float*)d_out;

  char* ws = (char*)d_ws;
  const size_t QK_ELEMS = (size_t)BATCH * NTOK * DQK;   // 131072 f32 each
  float*    Qf = (float*)ws;
  float*    Kf = Qf + QK_ELEMS;
  _Float16* Vh = (_Float16*)(ws + 2 * QK_ELEMS * sizeof(float)); // 2 MB

  qkv_proj_kernel<<<dim3((BATCH * NTOK) / 256), dim3(256), 0, stream>>>(
      x, wq, bq, wk, bk, wv, bv, Qf, Kf, Vh);

  attn_kernel<<<dim3(BATCH * (NTOK / 64)), dim3(128), 0, stream>>>(
      Qf, Kf, Vh, x, gamma, out);
}
